// FeaturePropogation_65910568124553
// MI455X (gfx1250) — matrix operand order repaired and checked by reading, not compiled, not run on hardware
//
#include <hip/hip_runtime.h>
#include <hip/hip_bf16.h>
#include <stdint.h>

// ---------------------------------------------------------------------------
// Problem constants (from reference setup_inputs)
// ---------------------------------------------------------------------------
#define BB      8
#define NN      16384
#define MM      4096
#define C1      64
#define C2      256
#define CO      256
#define K1      (C1 + C2)          // 320
#define K2      256
#define NCOL    (BB * NN)          // 131072 = 2^17
#define EPS_D   1e-8f
#define EPS_BN  1e-5f

typedef __attribute__((ext_vector_type(8)))  __bf16 v8bf;
typedef __attribute__((ext_vector_type(16))) __bf16 v16bf;
typedef __attribute__((ext_vector_type(8)))  float  v8f;

static __device__ __forceinline__ v16bf cat16(v8bf lo, v8bf hi) {
    return __builtin_shufflevector(lo, hi, 0,1,2,3,4,5,6,7,8,9,10,11,12,13,14,15);
}

// ---------------------------------------------------------------------------
// 0) zero the BN stats accumulators (1024 floats)
// ---------------------------------------------------------------------------
__global__ __launch_bounds__(256) void k_zero_stats(float* s) {
    s[blockIdx.x * 256 + threadIdx.x] = 0.0f;
}

// ---------------------------------------------------------------------------
// 1) fp32 -> bf16 weight conversion
// ---------------------------------------------------------------------------
__global__ __launch_bounds__(256) void k_w_to_bf16(const float* __restrict__ w,
                                                   __bf16* __restrict__ o, int n) {
    int i = blockIdx.x * 256 + threadIdx.x;
    if (i < n) o[i] = (__bf16)w[i];
}

// ---------------------------------------------------------------------------
// 2) 3-NN inverse-distance interpolation + concat -> X[ncol][K1] (bf16)
// ---------------------------------------------------------------------------
__global__ __launch_bounds__(256) void k_build_x(const float* __restrict__ p1,
                                                 const float* __restrict__ p2,
                                                 const float* __restrict__ f1,
                                                 const float* __restrict__ f2,
                                                 const int*   __restrict__ idx,
                                                 __bf16* __restrict__ X) {
    const int tid  = threadIdx.x;
    const int ncol = blockIdx.x * 256 + tid;
    const int b = ncol >> 14;          // /NN
    const int n = ncol & (NN - 1);

    const float px = p1[(size_t)ncol * 3 + 0];
    const float py = p1[(size_t)ncol * 3 + 1];
    const float pz = p1[(size_t)ncol * 3 + 2];

    const int i0 = idx[(size_t)ncol * 3 + 0];
    const int i1 = idx[(size_t)ncol * 3 + 1];
    const int i2 = idx[(size_t)ncol * 3 + 2];

    const float* p2b = p2 + (size_t)b * MM * 3;
    float dx, dy, dz;
    dx = px - p2b[i0*3+0]; dy = py - p2b[i0*3+1]; dz = pz - p2b[i0*3+2];
    float r0 = 1.0f / (dx*dx + dy*dy + dz*dz + EPS_D);
    dx = px - p2b[i1*3+0]; dy = py - p2b[i1*3+1]; dz = pz - p2b[i1*3+2];
    float r1 = 1.0f / (dx*dx + dy*dy + dz*dz + EPS_D);
    dx = px - p2b[i2*3+0]; dy = py - p2b[i2*3+1]; dz = pz - p2b[i2*3+2];
    float r2 = 1.0f / (dx*dx + dy*dy + dz*dz + EPS_D);
    const float rs = 1.0f / (r0 + r1 + r2);
    const float w0 = r0 * rs, w1 = r1 * rs, w2 = r2 * rs;

    __bf16* xrow = X + (size_t)ncol * K1;

    // f1 channels (coalesced across lanes per channel)
    const float* f1b = f1 + (size_t)b * C1 * NN + n;
    #pragma unroll
    for (int c8 = 0; c8 < C1; c8 += 8) {
        v8bf pk;
        #pragma unroll
        for (int j = 0; j < 8; ++j)
            pk[j] = (__bf16)f1b[(size_t)(c8 + j) * NN];
        *(v8bf*)(xrow + c8) = pk;
    }

    // interpolated f2 channels
    const float* f2b = f2 + (size_t)b * C2 * MM;
    const float* q0 = f2b + i0;
    const float* q1 = f2b + i1;
    const float* q2 = f2b + i2;
    #pragma unroll 4
    for (int c8 = 0; c8 < C2; c8 += 8) {
        v8bf pk;
        #pragma unroll
        for (int j = 0; j < 8; ++j) {
            const size_t co = (size_t)(c8 + j) * MM;
            pk[j] = (__bf16)(w0 * q0[co] + w1 * q1[co] + w2 * q2[co]);
        }
        *(v8bf*)(xrow + C1 + c8) = pk;
    }
}

// ---------------------------------------------------------------------------
// 3) bf16 WMMA GEMM + fused BN-stats:
//      Y[256][NCOL] = A[256][K] * X[NCOL][K]^T  (f32 accumulate)
//      sum[c] += sum_n Y[c][n] ; sq[c] += sum_n Y[c][n]^2   (atomic f32)
//    Wave tile = 32(m) x 64(n): 2 A fragments x 4 B fragments -> 8 WMMA
//    per 12 b128 lane-loads  (21.3 FLOP/B from L2; X stays L2-resident).
//    A fragment (ISA 16-bit A 16x32): lane m = lane&15, hs = lane>>4,
//      halves 0..7  = K[k0+hs*8 .. +7], halves 8..15 = K[k0+16+hs*8 .. +7]
//    B fragment (ISA B KxN): lane n = lane&15, halves 0..7 = K[k0+hs*8 .. +7], halves 8..15 = K[k0+16+hs*8 .. +7]
//    D layout: VGPR r -> row m = r + 8*hs, col n = lane&15
// ---------------------------------------------------------------------------
__global__ __launch_bounds__(256) void k_gemm_bf16(const __bf16* __restrict__ A,
                                                   const __bf16* __restrict__ X,
                                                   float* __restrict__ Y,
                                                   float* __restrict__ sum,
                                                   float* __restrict__ sq,
                                                   int K) {
    const int tid  = threadIdx.x;
    const int wave = tid >> 5;
    const int lane = tid & 31;
    const int lq   = lane & 15;
    const int hs   = lane >> 4;
    const int wm   = wave >> 2;                               // 0..1
    const int wn   = wave & 3;                                // 0..3

    const int mbase = blockIdx.y * 64 + wm * 32;              // 2 m-tiles
    const int nbase = blockIdx.x * 256 + wn * 64;             // 4 n-tiles

    const __bf16* Arow0 = A + (size_t)(mbase + lq) * K;
    const __bf16* Arow1 = A + (size_t)(mbase + 16 + lq) * K;

    v8f acc[4][2];
    #pragma unroll
    for (int g = 0; g < 4; ++g)
        #pragma unroll
        for (int t = 0; t < 2; ++t)
            #pragma unroll
            for (int r = 0; r < 8; ++r) acc[g][t][r] = 0.0f;

    for (int k0 = 0; k0 < K; k0 += 32) {
        const v16bf a0 = cat16(*(const v8bf*)(Arow0 + k0 + hs * 8),
                               *(const v8bf*)(Arow0 + k0 + 16 + hs * 8));
        const v16bf a1 = cat16(*(const v8bf*)(Arow1 + k0 + hs * 8),
                               *(const v8bf*)(Arow1 + k0 + 16 + hs * 8));
        #pragma unroll
        for (int g = 0; g < 4; ++g) {
            const int ncol = nbase + g * 16 + lq;
            const __bf16* xp = X + (size_t)ncol * K + k0 + hs * 8;
            const v16bf bb = cat16(*(const v8bf*)(xp), *(const v8bf*)(xp + 16));
            acc[g][0] = __builtin_amdgcn_wmma_f32_16x16x32_bf16(
                false, a0, false, bb, (short)0, acc[g][0], false, false);
            acc[g][1] = __builtin_amdgcn_wmma_f32_16x16x32_bf16(
                false, a1, false, bb, (short)0, acc[g][1], false, false);
        }
    }

    // store D + fused per-row sum / sumsq (reduced across the 16 lanes
    // holding a row; xor masks 1..8 never cross the 16-lane half)
    #pragma unroll
    for (int t = 0; t < 2; ++t) {
        #pragma unroll
        for (int r = 0; r < 8; ++r) {
            const int mr = mbase + t * 16 + r + 8 * hs;
            float s = 0.0f, q = 0.0f;
            #pragma unroll
            for (int g = 0; g < 4; ++g) {
                const float v = acc[g][t][r];
                Y[(size_t)mr * NCOL + nbase + g * 16 + lq] = v;
                s += v; q += v * v;
            }
            #pragma unroll
            for (int off = 1; off < 16; off <<= 1) {
                s += __shfl_xor(s, off, 32);
                q += __shfl_xor(q, off, 32);
            }
            if (lq == 0) {
                atomicAdd(&sum[mr], s);
                atomicAdd(&sq[mr],  q);
            }
        }
    }
}

// ---------------------------------------------------------------------------
// 4) BN + ReLU, write bf16 X2[ncol][256] via LDS transpose tile (32 cols)
// ---------------------------------------------------------------------------
__global__ __launch_bounds__(256) void k_bn_relu_bf16(const float* __restrict__ Y,
                                                      const float* __restrict__ sum,
                                                      const float* __restrict__ sq,
                                                      const float* __restrict__ gam,
                                                      const float* __restrict__ bet,
                                                      __bf16* __restrict__ X2) {
    __shared__ __bf16 sh[32][264];           // 264 = 256 + 8 pad (16B-aligned rows)
    const int tid   = threadIdx.x;
    const int ncol0 = blockIdx.x * 32;
    const int col   = tid & 31;
    const int mg    = tid >> 5;
    const float inv = 1.0f / (float)NCOL;

    #pragma unroll 4
    for (int k = 0; k < 32; ++k) {
        const int m = mg + k * 8;
        const float mean = sum[m] * inv;
        const float var  = sq[m] * inv - mean * mean;
        const float sc   = gam[m] * rsqrtf(var + EPS_BN);
        const float sf   = bet[m] - mean * sc;
        const float v    = Y[(size_t)m * NCOL + ncol0 + col] * sc + sf;
        sh[col][m] = (__bf16)(v > 0.0f ? v : 0.0f);
    }
    __syncthreads();

    const int oc = tid >> 3, chunk = tid & 7;
    __bf16* dst = X2 + (size_t)(ncol0 + oc) * 256 + chunk * 32;
    const v8bf* src = (const v8bf*)&sh[oc][chunk * 32];
    #pragma unroll
    for (int j = 0; j < 4; ++j) ((v8bf*)dst)[j] = src[j];
}

// ---------------------------------------------------------------------------
// 5) final BN + ReLU, write fp32 out[B][256][N]
// ---------------------------------------------------------------------------
__global__ __launch_bounds__(256) void k_bn_relu_out(const float* __restrict__ Y,
                                                     const float* __restrict__ sum,
                                                     const float* __restrict__ sq,
                                                     const float* __restrict__ gam,
                                                     const float* __restrict__ bet,
                                                     float* __restrict__ out) {
    const size_t i = (size_t)blockIdx.x * 256 + threadIdx.x;   // over 256*NCOL
    const int c    = (int)(i >> 17);           // / NCOL
    const int ncol = (int)(i & (NCOL - 1));
    const int b    = ncol >> 14;               // / NN
    const int n    = ncol & (NN - 1);
    const float inv  = 1.0f / (float)NCOL;
    const float mean = sum[c] * inv;
    const float var  = sq[c] * inv - mean * mean;
    const float sc   = gam[c] * rsqrtf(var + EPS_BN);
    const float sf   = bet[c] - mean * sc;
    const float v    = Y[i] * sc + sf;
    out[((size_t)(b * CO + c)) * NN + n] = v > 0.0f ? v : 0.0f;
}

// ---------------------------------------------------------------------------
// launch
// ---------------------------------------------------------------------------
extern "C" void kernel_launch(void* const* d_in, const int* in_sizes, int n_in,
                              void* d_out, int out_size, void* d_ws, size_t ws_size,
                              hipStream_t stream) {
    const float* p1  = (const float*)d_in[0];
    const float* p2  = (const float*)d_in[1];
    const float* f1  = (const float*)d_in[2];
    const float* f2  = (const float*)d_in[3];
    const int*   idx = (const int*)  d_in[4];
    const float* W1  = (const float*)d_in[5];
    const float* g1  = (const float*)d_in[6];
    const float* b1  = (const float*)d_in[7];
    const float* W2  = (const float*)d_in[8];
    const float* g2  = (const float*)d_in[9];
    const float* b2  = (const float*)d_in[10];
    float* out = (float*)d_out;

    // workspace partition (X2 aliases X1: X1 dead after GEMM1)
    char* ws = (char*)d_ws;
    const size_t offX  = 0;                                   // 131072*320*2  = 80 MiB
    const size_t offY  = offX + (size_t)NCOL * K1 * 2;        // 131072*256*4  = 128 MiB
    const size_t offW1 = offY + (size_t)NCOL * CO * 4;        // 256*320*2
    const size_t offW2 = offW1 + (size_t)CO * K1 * 2;         // 256*256*2
    const size_t offS  = offW2 + (size_t)CO * K2 * 2;         // 4*256 floats

    __bf16* X1  = (__bf16*)(ws + offX);
    __bf16* X2  = (__bf16*)(ws + offX);                       // alias
    float*  Y   = (float*) (ws + offY);
    __bf16* Wb1 = (__bf16*)(ws + offW1);
    __bf16* Wb2 = (__bf16*)(ws + offW2);
    float*  S   = (float*) (ws + offS);
    float *sum1 = S, *sq1 = S + 256, *sum2 = S + 512, *sq2 = S + 768;

    // 0) zero stats
    k_zero_stats<<<4, 256, 0, stream>>>(S);
    // 1) weights to bf16
    k_w_to_bf16<<<(CO * K1 + 255) / 256, 256, 0, stream>>>(W1, Wb1, CO * K1);
    k_w_to_bf16<<<(CO * K2 + 255) / 256, 256, 0, stream>>>(W2, Wb2, CO * K2);
    // 2) interpolation + concat -> X1
    k_build_x<<<NCOL / 256, 256, 0, stream>>>(p1, p2, f1, f2, idx, X1);
    // 3) GEMM1 (+ fused BN stats 1): Y = W1 * X1
    k_gemm_bf16<<<dim3(NCOL / 256, CO / 64), 256, 0, stream>>>(Wb1, X1, Y, sum1, sq1, K1);
    // 4) BN+ReLU -> X2 (bf16, transposed layout)
    k_bn_relu_bf16<<<NCOL / 32, 256, 0, stream>>>(Y, sum1, sq1, g1, b1, X2);
    // 5) GEMM2 (+ fused BN stats 2): Y = W2 * X2
    k_gemm_bf16<<<dim3(NCOL / 256, CO / 64), 256, 0, stream>>>(Wb2, X2, Y, sum2, sq2, K2);
    // 6) final BN+ReLU -> out [B,256,N]
    k_bn_relu_out<<<(size_t)CO * NCOL / 256, 256, 0, stream>>>(Y, sum2, sq2, g2, b2, out);
}
